// TreeDecoderTeacherForced_16458314678334
// MI455X (gfx1250) — matrix operand order repaired and checked by reading, not compiled, not run on hardware
//
#include <hip/hip_runtime.h>
#include <hip/hip_bf16.h>

// ---------------------------------------------------------------------------
// TreeDecoderTeacherForced on MI455X (gfx1250, wave32, WMMA bf16)
// D=9 levels, H=E=64, POS=39 (padded to 192 for the fuse GEMM K-dim).
// All activations kept in bf16; all matmuls via v_wmma_f32_16x16x32_bf16.
// Weights pre-packed into WMMA B-fragment layout (512 bf16 per 32x16 tile).
// ---------------------------------------------------------------------------

#define HDIM 64
#define EDIM 64
#define XROW 192   // padded fuse input dim (167 -> 192 = 6 K-tiles of 32)

typedef __bf16 v16bf __attribute__((ext_vector_type(16)));
typedef __bf16 v8bf  __attribute__((ext_vector_type(8)));
typedef float  v8f   __attribute__((ext_vector_type(8)));

// fragment-table offsets (units of 512-element fragments)
#define FRAG_FUSE_W1   0   // 6x4  = 24 frags (K=192, N=64)
#define FRAG_FUSE_W2  24   // 2x4  = 8
#define FRAG_VAL_W1   32   // 2x4  = 8
#define FRAG_SPLIT_W1 40   // 2x4  = 8
#define FRAG_CHILD_W1 48   // 2x4  = 8
#define FRAG_CHILD_W2 56   // 2x16 = 32  (K=64, N=256)
#define FRAG_MIX      88   // 18x4 = 72 per level (d-1), levels 1..9

// ---------------------------------------------------------------------------
// WMMA helpers (ISA 7.12 layouts, wave32)
// ---------------------------------------------------------------------------
__device__ __forceinline__ v8f wmma_bf16(v16bf a, v16bf b, v8f c) {
  return __builtin_amdgcn_wmma_f32_16x16x32_bf16(false, a, false, b,
                                                 (short)0, c, false, false);
}

__device__ __forceinline__ v16bf bf16_zero_frag() {
  union { v16bf v; __bf16 e[16]; } u;
#pragma unroll
  for (int i = 0; i < 16; ++i) u.e[i] = (__bf16)0.0f;
  return u.v;
}

// A fragment (16x32 bf16, MxK) from a row-major matrix, row = m0 + (lane&15).
// ISA layout: lanes 0-15 hold K=[8h.. ] chunks; VGPR0-3: K=8h..8h+7,
// VGPR4-7: K=16+8h..16+8h+7 where h = lane>>4. Two aligned b128 loads.
__device__ __forceinline__ v16bf load_a_rowmajor(const __bf16* base, int ld,
                                                 int m0, int kbase, int lane) {
  int M = lane & 15, h = lane >> 4;
  const __bf16* row = base + (size_t)(m0 + M) * ld + kbase + 8 * h;
  union { v16bf v; v8bf q[2]; } u;
  u.q[0] = *(const v8bf*)(row);
  u.q[1] = *(const v8bf*)(row + 16);
  return u.v;
}

__device__ __forceinline__ v16bf load_a_rowmajor_clamp(const __bf16* base,
                                                       int ld, int m0,
                                                       int kbase, int lane,
                                                       int n) {
  int M = lane & 15, h = lane >> 4;
  int r = m0 + M; if (r > n - 1) r = n - 1;
  const __bf16* row = base + (size_t)r * ld + kbase + 8 * h;
  union { v16bf v; v8bf q[2]; } u;
  u.q[0] = *(const v8bf*)(row);
  u.q[1] = *(const v8bf*)(row + 16);
  return u.v;
}

// A fragment where each lane supplies its own row pointer (neighbor gather).
__device__ __forceinline__ v16bf load_a_row(const __bf16* row, int kbase,
                                            int lane) {
  int h = lane >> 4;
  const __bf16* p = row + kbase + 8 * h;
  union { v16bf v; v8bf q[2]; } u;
  u.q[0] = *(const v8bf*)(p);
  u.q[1] = *(const v8bf*)(p + 16);
  return u.v;
}

// B fragment from the pre-packed table: lane-major, 16 contiguous bf16/lane.
__device__ __forceinline__ v16bf load_b_frag(const __bf16* packW, int frag,
                                             int lane) {
  const __bf16* p = packW + (size_t)frag * 512 + lane * 16;
  union { v16bf v; v8bf q[2]; } u;
  u.q[0] = *(const v8bf*)(p);
  u.q[1] = *(const v8bf*)(p + 8);
  return u.v;
}

// D tile (16x16 f32): lane -> N = lane&15, VGPR i -> M = i + 8*(lane>>4).
__device__ __forceinline__ void store_d_lds_bf16(v8f acc, __bf16* dst, int ld,
                                                 int m0, int n0,
                                                 const float* bias, bool relu,
                                                 int lane) {
  int nloc = lane & 15, h = lane >> 4;
  float b = bias[n0 + nloc];
#pragma unroll
  for (int i = 0; i < 8; ++i) {
    float v = acc[i] + b;
    if (relu) v = fmaxf(v, 0.0f);
    dst[(size_t)(m0 + i + 8 * h) * ld + n0 + nloc] = (__bf16)v;
  }
}

__device__ __forceinline__ void store_d_lds_f32(v8f acc, float* dst, int ld,
                                                int m0, int n0,
                                                const float* bias, bool relu,
                                                int lane) {
  int nloc = lane & 15, h = lane >> 4;
  float b = bias[n0 + nloc];
#pragma unroll
  for (int i = 0; i < 8; ++i) {
    float v = acc[i] + b;
    if (relu) v = fmaxf(v, 0.0f);
    dst[(size_t)(m0 + i + 8 * h) * ld + n0 + nloc] = v;
  }
}

__device__ __forceinline__ void store_d_global_bf16(v8f acc, __bf16* base,
                                                    int m0_abs, int n0,
                                                    const float* bias,
                                                    bool relu, int lane,
                                                    int n) {
  int nloc = lane & 15, h = lane >> 4;
  float b = bias[n0 + nloc];
#pragma unroll
  for (int i = 0; i < 8; ++i) {
    int gr = m0_abs + i + 8 * h;
    if (gr < n) {
      float v = acc[i] + b;
      if (relu) v = fmaxf(v, 0.0f);
      base[(size_t)gr * HDIM + n0 + nloc] = (__bf16)v;
    }
  }
}

__device__ __forceinline__ unsigned dein(unsigned x) {
  x &= 0x55555555u;
  x = (x | (x >> 1)) & 0x33333333u;
  x = (x | (x >> 2)) & 0x0F0F0F0Fu;
  x = (x | (x >> 4)) & 0x00FF00FFu;
  x = (x | (x >> 8)) & 0x0000FFFFu;
  return x;
}

// ---------------------------------------------------------------------------
// Weight packing: f32 (Ksrc x N, row-major) -> bf16 B-fragments.
// Fragment (kt,nt): lane L holds N = nt*16 + (L&15), K-half = L>>4;
// slot v (0..15): K = kt*32 + 8*(L>>4) + v (v<8), kt*32 + 16 + 8*(L>>4) + (v-8) (v>=8). Out-of-range K/N -> 0.
// ---------------------------------------------------------------------------
__global__ void pack_b_kernel(const float* __restrict__ W,
                              __bf16* __restrict__ dst, int Ksrc, int N,
                              int NT) {
  int frag = blockIdx.x;
  int kt = frag / NT, nt = frag % NT;
  int t = threadIdx.x;            // 0..511
  int lane = t >> 4, v = t & 15;
  int ncol = nt * 16 + (lane & 15);
  int k = kt * 32 + (lane >> 4) * 8 + (v < 8 ? v : 16 + (v - 8));
  float val = (k < Ksrc && ncol < N) ? W[(size_t)k * N + ncol] : 0.0f;
  dst[(size_t)frag * 512 + lane * 16 + v] = (__bf16)val;
}

__global__ void init_root_kernel(const float* __restrict__ root,
                                 __bf16* __restrict__ h_prev) {
  int t = threadIdx.x;
  if (t < HDIM) h_prev[t] = (__bf16)root[t];
}

__global__ void zero_bf16_kernel(__bf16* __restrict__ p, int count) {
  int i = blockIdx.x * blockDim.x + threadIdx.x;
  if (i < count) p[i] = (__bf16)0.0f;
}

// ---------------------------------------------------------------------------
// Fuse: x = [h_prev | LN(emb) | fourier(pos)] (padded to 192) ->
//       h = (relu(x@W1+b1))@W2+b2.  Block = 128 thr (4 waves) = 64 rows.
// ---------------------------------------------------------------------------
__global__ __launch_bounds__(128) void fuse_kernel(
    const __bf16* __restrict__ h_prev, const float* __restrict__ emb,
    const int* __restrict__ keys, const float* __restrict__ skip_g,
    const float* __restrict__ skip_b, const float* __restrict__ fuse_b1,
    const float* __restrict__ fuse_b2, const __bf16* __restrict__ packW,
    __bf16* __restrict__ h_out, int n, float res, float zlev) {
  __shared__ __bf16 xs[64 * XROW];   // 24 KB
  __shared__ __bf16 ts[64 * HDIM];   //  8 KB
  int tid = threadIdx.x;
  int r0 = blockIdx.x * 64;

  // ---- Phase A: build input rows ----
  if (tid < 64) {
    int rr = tid;
    int gr = r0 + rr; if (gr > n - 1) gr = n - 1;
    __bf16* row = xs + (size_t)rr * XROW;
    const __bf16* hp = h_prev + (size_t)gr * HDIM;
#pragma unroll 8
    for (int j = 0; j < HDIM; ++j) row[j] = hp[j];
    // layernorm of emb row
    const float* er = emb + (size_t)gr * EDIM;
    float mu = 0.0f;
    for (int j = 0; j < EDIM; ++j) mu += er[j];
    mu *= (1.0f / EDIM);
    float var = 0.0f;
    for (int j = 0; j < EDIM; ++j) { float d = er[j] - mu; var += d * d; }
    var *= (1.0f / EDIM);
    float inv = rsqrtf(var + 1e-5f);
    for (int j = 0; j < EDIM; ++j)
      row[HDIM + j] = (__bf16)((er[j] - mu) * inv * skip_g[j] + skip_b[j]);
  } else {
    int rr = tid - 64;
    int gr = r0 + rr; if (gr > n - 1) gr = n - 1;
    __bf16* row = xs + (size_t)rr * XROW;
    unsigned key = (unsigned)keys[gr];
    float px = ((float)dein(key) + 0.5f) / res;
    float py = ((float)dein(key >> 1) + 0.5f) / res;
    float pc[3] = {px, py, zlev};
    row[128] = (__bf16)px; row[129] = (__bf16)py; row[130] = (__bf16)zlev;
    int o = 131;
#pragma unroll
    for (int c = 0; c < 3; ++c) {
      float base = 6.28318530717958647692f * pc[c];
      float fr = 1.0f;
#pragma unroll
      for (int f = 0; f < 6; ++f) {
        row[o + f]     = (__bf16)__sinf(base * fr);
        row[o + 6 + f] = (__bf16)__cosf(base * fr);
        fr *= 2.0f;
      }
      o += 12;
    }
    for (int k = 167; k < XROW; ++k) row[k] = (__bf16)0.0f;
  }
  __syncthreads();

  int wv = tid >> 5, lane = tid & 31;
  int m0 = wv * 16;

  // ---- Phase B: t = relu(x @ W1 + b1) ----
#pragma unroll
  for (int nt = 0; nt < 4; ++nt) {
    v8f acc = {};
#pragma unroll
    for (int kt = 0; kt < 6; ++kt) {
      v16bf a = load_a_rowmajor(xs, XROW, m0, kt * 32, lane);
      v16bf b = load_b_frag(packW, FRAG_FUSE_W1 + kt * 4 + nt, lane);
      acc = wmma_bf16(a, b, acc);
    }
    store_d_lds_bf16(acc, ts, HDIM, m0, nt * 16, fuse_b1, true, lane);
  }
  __syncthreads();

  // ---- Phase C: h = t @ W2 + b2 ----
#pragma unroll
  for (int nt = 0; nt < 4; ++nt) {
    v8f acc = {};
#pragma unroll
    for (int kt = 0; kt < 2; ++kt) {
      v16bf a = load_a_rowmajor(ts, HDIM, m0, kt * 32, lane);
      v16bf b = load_b_frag(packW, FRAG_FUSE_W2 + kt * 4 + nt, lane);
      acc = wmma_bf16(a, b, acc);
    }
    store_d_global_bf16(acc, h_out, r0 + m0, nt * 16, fuse_b2, false, lane, n);
  }
}

// ---------------------------------------------------------------------------
// Mix (d>=1): h = relu(gather9(h) @ mix_w[d] + mix_b[d]).
// K = 576 = 9 neighbors x 64; A fragments gathered straight from neighbor
// rows of h_fuse (no n x 576 materialization -> saves ~300 MB at d=9).
// ---------------------------------------------------------------------------
__global__ __launch_bounds__(128) void mix_kernel(
    const __bf16* __restrict__ h_in, const int* __restrict__ neigh,
    const __bf16* __restrict__ packMix, const float* __restrict__ mix_b,
    __bf16* __restrict__ h_out, int n) {
  int tid = threadIdx.x, wv = tid >> 5, lane = tid & 31;
  int r0 = blockIdx.x * 64, m0 = wv * 16;
  int myrow = r0 + m0 + (lane & 15); if (myrow > n - 1) myrow = n - 1;
  int ng[9];
#pragma unroll
  for (int j = 0; j < 9; ++j) ng[j] = neigh[(size_t)myrow * 9 + j];

#pragma unroll
  for (int nt = 0; nt < 4; ++nt) {
    v8f acc = {};
#pragma unroll 2
    for (int kt = 0; kt < 18; ++kt) {
      int g = ng[kt >> 1];
      v16bf a;
      if (g >= 0) a = load_a_row(h_in + (size_t)g * HDIM, (kt & 1) * 32, lane);
      else        a = bf16_zero_frag();
      v16bf b = load_b_frag(packMix, kt * 4 + nt, lane);
      acc = wmma_bf16(a, b, acc);
    }
    store_d_global_bf16(acc, h_out, r0 + m0, nt * 16, mix_b, true, lane, n);
  }
}

// ---------------------------------------------------------------------------
// Heads: val (all d), split + child-scatter (d<D).
// ---------------------------------------------------------------------------
__global__ __launch_bounds__(128) void heads_kernel(
    const __bf16* __restrict__ h_in, const __bf16* __restrict__ packW,
    const float* __restrict__ val_b1, const float* __restrict__ val_w2,
    const float* __restrict__ val_b2, const float* __restrict__ split_b1,
    const float* __restrict__ split_w2, const float* __restrict__ split_b2,
    const float* __restrict__ child_b1, const float* __restrict__ child_b2,
    const int* __restrict__ children, float* __restrict__ out,
    __bf16* __restrict__ h_next, int n, int n_next, int out_val_off,
    int out_split_off, int do_child) {
  __shared__ float  tf[64 * HDIM];   // 16 KB (head hidden, f32)
  __shared__ __bf16 tc[64 * HDIM];   //  8 KB (child hidden, bf16)
  int tid = threadIdx.x, wv = tid >> 5, lane = tid & 31;
  int r0 = blockIdx.x * 64, m0 = wv * 16;

  // ---- val hidden: relu(h @ val_w1 + b1) -> tf ----
#pragma unroll
  for (int nt = 0; nt < 4; ++nt) {
    v8f acc = {};
#pragma unroll
    for (int kt = 0; kt < 2; ++kt) {
      v16bf a = load_a_rowmajor_clamp(h_in, HDIM, r0 + m0, kt * 32, lane, n);
      v16bf b = load_b_frag(packW, FRAG_VAL_W1 + kt * 4 + nt, lane);
      acc = wmma_bf16(a, b, acc);
    }
    store_d_lds_f32(acc, tf, HDIM, m0, nt * 16, val_b1, true, lane);
  }
  __syncthreads();
  if (tid < 64) {
    int gr = r0 + tid;
    if (gr < n) {
      float s = val_b2[0];
      const float* row = tf + (size_t)tid * HDIM;
#pragma unroll 8
      for (int c = 0; c < HDIM; ++c) s += row[c] * val_w2[c];
      out[(size_t)out_val_off + gr] = s;
    }
  }
  __syncthreads();

  if (do_child) {
    // ---- split hidden + reduce ----
#pragma unroll
    for (int nt = 0; nt < 4; ++nt) {
      v8f acc = {};
#pragma unroll
      for (int kt = 0; kt < 2; ++kt) {
        v16bf a = load_a_rowmajor_clamp(h_in, HDIM, r0 + m0, kt * 32, lane, n);
        v16bf b = load_b_frag(packW, FRAG_SPLIT_W1 + kt * 4 + nt, lane);
        acc = wmma_bf16(a, b, acc);
      }
      store_d_lds_f32(acc, tf, HDIM, m0, nt * 16, split_b1, true, lane);
    }
    __syncthreads();
    if (tid < 64) {
      int gr = r0 + tid;
      if (gr < n) {
        float s = split_b2[0];
        const float* row = tf + (size_t)tid * HDIM;
#pragma unroll 8
        for (int c = 0; c < HDIM; ++c) s += row[c] * split_w2[c];
        out[(size_t)out_split_off + gr] = s;
      }
    }
    __syncthreads();

    // ---- child hidden: relu(h @ child_w1 + b1) -> tc (bf16) ----
#pragma unroll
    for (int nt = 0; nt < 4; ++nt) {
      v8f acc = {};
#pragma unroll
      for (int kt = 0; kt < 2; ++kt) {
        v16bf a = load_a_rowmajor_clamp(h_in, HDIM, r0 + m0, kt * 32, lane, n);
        v16bf b = load_b_frag(packW, FRAG_CHILD_W1 + kt * 4 + nt, lane);
        acc = wmma_bf16(a, b, acc);
      }
      store_d_lds_bf16(acc, tc, HDIM, m0, nt * 16, child_b1, true, lane);
    }
    __syncthreads();

    // ---- child out: cf = tc @ child_w2 + b2 (N=256), scatter to h_next ----
#pragma unroll 4
    for (int nt = 0; nt < 16; ++nt) {
      v8f acc = {};
#pragma unroll
      for (int kt = 0; kt < 2; ++kt) {
        v16bf a = load_a_rowmajor(tc, HDIM, m0, kt * 32, lane);
        v16bf b = load_b_frag(packW, FRAG_CHILD_W2 + kt * 16 + nt, lane);
        acc = wmma_bf16(a, b, acc);
      }
      int nloc = lane & 15, hh = lane >> 4;
      int col = nt * 16 + nloc;
      int c4 = col >> 6, f = col & 63;
      float b2 = child_b2[col];
#pragma unroll
      for (int i = 0; i < 8; ++i) {
        int gr = r0 + m0 + i + 8 * hh;
        if (gr < n) {
          int ch = children[(size_t)gr * 4 + c4];
          if (ch >= 0 && ch < n_next)
            h_next[(size_t)ch * HDIM + f] = (__bf16)(acc[i] + b2);
        }
      }
    }
  }
}

// ---------------------------------------------------------------------------
// Host launch
// ---------------------------------------------------------------------------
static constexpr int kSizes[10] = {1, 4, 16, 64, 256, 1024, 4096, 16384,
                                   65536, 262144};
static constexpr int kOff[11] = {0, 1, 5, 21, 85, 341, 1365, 5461, 21845,
                                 87381, 349525};

extern "C" void kernel_launch(void* const* d_in, const int* in_sizes, int n_in,
                              void* d_out, int out_size, void* d_ws,
                              size_t ws_size, hipStream_t stream) {
  const int*   keys     = (const int*)d_in[0];
  const int*   neighs   = (const int*)d_in[1];
  const int*   children = (const int*)d_in[2];
  const float* emb      = (const float*)d_in[3];
  const float* root     = (const float*)d_in[4];
  const float* skip_g   = (const float*)d_in[5];
  const float* skip_b   = (const float*)d_in[6];
  const float* fuse_w1  = (const float*)d_in[7];
  const float* fuse_b1  = (const float*)d_in[8];
  const float* fuse_w2  = (const float*)d_in[9];
  const float* fuse_b2  = (const float*)d_in[10];
  const float* split_w1 = (const float*)d_in[11];
  const float* split_b1 = (const float*)d_in[12];
  const float* split_w2 = (const float*)d_in[13];
  const float* split_b2 = (const float*)d_in[14];
  const float* child_w1 = (const float*)d_in[15];
  const float* child_b1 = (const float*)d_in[16];
  const float* child_w2 = (const float*)d_in[17];
  const float* child_b2 = (const float*)d_in[18];
  const float* val_w1   = (const float*)d_in[19];
  const float* val_b1   = (const float*)d_in[20];
  const float* val_w2   = (const float*)d_in[21];
  const float* val_b2   = (const float*)d_in[22];
  const float* mix_w    = (const float*)d_in[23];
  const float* mix_b    = (const float*)d_in[24];
  float* out = (float*)d_out;

  // workspace layout: [packed weights 1MB][h_prev 32MB][h_fuse 32MB][h_mix 32MB]
  __bf16* packW  = (__bf16*)d_ws;
  __bf16* h_prev = (__bf16*)((char*)d_ws + (1 << 20));
  __bf16* h_fuse = h_prev + (size_t)262144 * HDIM;
  __bf16* h_mix  = h_fuse + (size_t)262144 * HDIM;

  // ---- pack all weights into WMMA B-fragment layout (bf16) ----
  pack_b_kernel<<<24, 512, 0, stream>>>(fuse_w1,
      packW + (size_t)FRAG_FUSE_W1 * 512, 167, 64, 4);
  pack_b_kernel<<<8, 512, 0, stream>>>(fuse_w2,
      packW + (size_t)FRAG_FUSE_W2 * 512, 64, 64, 4);
  pack_b_kernel<<<8, 512, 0, stream>>>(val_w1,
      packW + (size_t)FRAG_VAL_W1 * 512, 64, 64, 4);
  pack_b_kernel<<<8, 512, 0, stream>>>(split_w1,
      packW + (size_t)FRAG_SPLIT_W1 * 512, 64, 64, 4);
  pack_b_kernel<<<8, 512, 0, stream>>>(child_w1,
      packW + (size_t)FRAG_CHILD_W1 * 512, 64, 64, 4);
  pack_b_kernel<<<32, 512, 0, stream>>>(child_w2,
      packW + (size_t)FRAG_CHILD_W2 * 512, 64, 256, 16);
  for (int d = 1; d <= 9; ++d)
    pack_b_kernel<<<72, 512, 0, stream>>>(mix_w + (size_t)d * 576 * 64,
        packW + (size_t)(FRAG_MIX + (d - 1) * 72) * 512, 576, 64, 4);

  init_root_kernel<<<1, 64, 0, stream>>>(root, h_prev);

  // ---- level pipeline (sequential dependency over levels) ----
  for (int d = 0; d <= 9; ++d) {
    int n = kSizes[d];
    int blocks = (n + 63) / 64;

    fuse_kernel<<<blocks, 128, 0, stream>>>(
        h_prev, emb + (size_t)kOff[d] * EDIM, keys + kOff[d], skip_g, skip_b,
        fuse_b1, fuse_b2, packW, h_fuse, n, (float)(1 << d),
        (float)d / 9.0f);

    const __bf16* hsel = h_fuse;
    if (d >= 1) {
      mix_kernel<<<blocks, 128, 0, stream>>>(
          h_fuse, neighs + (size_t)(kOff[d] - 1) * 9,
          packW + (size_t)(FRAG_MIX + (d - 1) * 72) * 512, mix_b + d * HDIM,
          h_mix, n);
      hsel = h_mix;
    }

    int n_next = (d < 9) ? kSizes[d + 1] : 0;
    if (d < 9) {
      int cnt = n_next * HDIM;
      zero_bf16_kernel<<<(cnt + 255) / 256, 256, 0, stream>>>(h_prev, cnt);
    }

    heads_kernel<<<blocks, 128, 0, stream>>>(
        hsel, packW, val_b1, val_w2, val_b2, split_b1, split_w2, split_b2,
        child_b1, child_b2, children + (size_t)kOff[d] * 4, out, h_prev, n,
        n_next, 87381 + kOff[d], kOff[d], (d < 9) ? 1 : 0);
  }
  (void)in_sizes; (void)n_in; (void)out_size; (void)ws_size;
}
